// AttentionLayer_21844203667667
// MI455X (gfx1250) — matrix operand order repaired and checked by reading, not compiled, not run on hardware
//
#include <hip/hip_runtime.h>
#include <hip/hip_bf16.h>
#include <math.h>

// ---------------------------------------------------------------------------
// MHA layer for MI455X (gfx1250), wave32, WMMA f32_16x16x32_f16.
// B=2, S=2048, D=1024, H=16, Dh=64. Outputs: out (B,S,D) f32 then attn
// (B,H,S,S) f32 concatenated in d_out.
//
// Attention is flash-style (register-resident online softmax, two passes over
// key tiles) so occupancy is VGPR-limited, not LDS-limited; K is stored
// transposed per head so score B-fragments read [d][s] row-major, lanes coalesced along s.
// ---------------------------------------------------------------------------

typedef __attribute__((ext_vector_type(16))) _Float16 v16h;
typedef __attribute__((ext_vector_type(8)))  _Float16 h8;
typedef __attribute__((ext_vector_type(8)))  float    v8f;

#define DEV static __device__ __forceinline__

// --- WMMA fragment index maps (per CDNA5 ISA 7.12.2) -----------------------
// A (16x32 f16, MxK): lane L holds row M=(L&15); VGPR v (pair sub=0/1):
//   k = (v>=4 ? 16:0) + ((L>>4)*8) + (v&3)*2 + sub
DEV int a_k_of(int lane, int v) {
  return ((v & 4) << 2) + ((lane >> 4) << 3) + ((v & 3) << 1);
}

// A fragment from a row pointer (rowp = &A[row*ld]), f32 or f16 source.
template <typename T>
DEV v16h load_a_frag_row(const T* rowp, int kbase, int lane) {
  v16h a;
#pragma unroll
  for (int v = 0; v < 8; ++v) {
    int k = kbase + a_k_of(lane, v);
    a[2 * v]     = (_Float16)rowp[k];
    a[2 * v + 1] = (_Float16)rowp[k + 1];
  }
  return a;
}

// B (32x16 f16, KxN) from row-major memory: lane holds N column col0+(lane&15),
// element e at k = kbase + (e>=8 ? 16:0) + ((lane>>4)*8) + (e&7).
DEV v16h load_b_frag_rm(const _Float16* B, int kbase, int ldn, int col0, int lane) {
  const _Float16* p = B + col0 + (lane & 15);
  v16h b;
#pragma unroll
  for (int e = 0; e < 16; ++e) {
    int k = kbase + ((e & 8) << 1) + ((lane >> 4) << 3) + (e & 7);
    b[e] = p[(size_t)k * ldn];
  }
  return b;
}

DEV v8f wmma_f16(v16h a, v16h b, v8f c) {
  return __builtin_amdgcn_wmma_f32_16x16x32_f16(false, a, false, b, (short)0, c,
                                                false, false);
}

DEV v8f vzero8() {
  v8f z = {0.f, 0.f, 0.f, 0.f, 0.f, 0.f, 0.f, 0.f};
  return z;
}

#define MASK_NEG (-1.0e30f)

// --- f32 -> f16 conversion --------------------------------------------------
__global__ void cvt_f32_f16(const float* __restrict__ in, _Float16* __restrict__ out,
                            int n) {
  int i = blockIdx.x * blockDim.x + threadIdx.x;
  if (i < n) out[i] = (_Float16)in[i];
}

// --- Generic WMMA GEMM: C(MxN) = A(MxK) @ B(KxN), one wave per 16x64 strip -
template <typename TA, bool OUT16>
__global__ __launch_bounds__(32) void wmma_gemm(const TA* __restrict__ A,
                                                const _Float16* __restrict__ B,
                                                void* __restrict__ Cout, int M,
                                                int N, int K) {
  const int lane = threadIdx.x;
  const int n0 = blockIdx.x * 64;
  const int m0 = blockIdx.y * 16;

  v8f acc[4];
#pragma unroll
  for (int j = 0; j < 4; ++j) acc[j] = vzero8();

  const TA* rowp = A + (size_t)(m0 + (lane & 15)) * K;
  for (int kb = 0; kb < K; kb += 32) {
    v16h a = load_a_frag_row(rowp, kb, lane);
#pragma unroll
    for (int j = 0; j < 4; ++j) {
      v16h b = load_b_frag_rm(B, kb, N, n0 + j * 16, lane);
      acc[j] = wmma_f16(a, b, acc[j]);
    }
  }

  const int mbase = m0 + ((lane >> 4) << 3);
#pragma unroll
  for (int j = 0; j < 4; ++j) {
    int nc = n0 + j * 16 + (lane & 15);
#pragma unroll
    for (int r = 0; r < 8; ++r) {
      size_t idx = (size_t)(mbase + r) * N + nc;
      if (OUT16)
        ((_Float16*)Cout)[idx] = (_Float16)acc[j][r];
      else
        ((float*)Cout)[idx] = acc[j][r];
    }
  }
}

// --- RoPE: Q rotated in place (row-major B*S x D); K rotated and written
// transposed per head: Kt[((b*H+h)*Dh + d)*S + s].
__global__ void rope_kernel(_Float16* __restrict__ Q,
                            const _Float16* __restrict__ Krm,
                            _Float16* __restrict__ Kt, int Bn, int S, int D,
                            int H, int Dh) {
  int idx = blockIdx.x * blockDim.x + threadIdx.x;
  int half = Dh >> 1;
  int total = Bn * S * H * half;
  if (idx >= total) return;
  int i = idx % half;
  int h = (idx / half) % H;
  int s = (idx / (half * H)) % S;
  int b = idx / (half * H * S);

  float inv_freq = __powf(10000.0f, -((float)(2 * i) / (float)Dh));
  float ang = (float)s * inv_freq;
  float c = __cosf(ang), sn = __sinf(ang);

  size_t base = ((size_t)(b * S + s)) * D + h * Dh + 2 * i;
  float q0 = (float)Q[base], q1 = (float)Q[base + 1];
  Q[base]     = (_Float16)(q0 * c - q1 * sn);
  Q[base + 1] = (_Float16)(q0 * sn + q1 * c);

  float k0 = (float)Krm[base], k1 = (float)Krm[base + 1];
  size_t tbase = ((size_t)(b * H + h) * Dh + 2 * i) * S + s;
  Kt[tbase]     = (_Float16)(k0 * c - k1 * sn);
  Kt[tbase + S] = (_Float16)(k0 * sn + k1 * c);
}

// --- Flash-style attention: one wave per (b, h, 16 query rows). ------------
// Pass 1: online row max/sum over all key tiles (registers only).
// Pass 2: recompute scores, normalize, store attn (f32), and accumulate
//         P @ V via a 1 KB LDS transpose tile (16x32 f16).
__global__ __launch_bounds__(32) void attn_kernel(
    const _Float16* __restrict__ Qh, const _Float16* __restrict__ Kt,
    const _Float16* __restrict__ Vh, const int* __restrict__ mask,
    float* __restrict__ attn_out, _Float16* __restrict__ AV, int Bn, int H,
    int S, int D, int Dh, float scale) {
  __shared__ _Float16 lp[16 * 32];  // P transpose tile

  const int lane = threadIdx.x;
  const int q0 = blockIdx.x * 16;
  const int h = blockIdx.y;
  const int b = blockIdx.z;
  const int mb = (lane >> 4) << 3;  // row offset of this lane's D-fragment
  const int nl = lane & 15;         // column of this lane's D-fragment

  const _Float16* Qb  = Qh + (size_t)b * S * D + h * Dh;           // [s][d]
  const _Float16* Ktb = Kt + ((size_t)b * H + h) * Dh * S;         // [d][s]
  const _Float16* Vb  = Vh + (size_t)b * S * D + h * Dh;           // [s][d]
  const int* mrow = mask + (size_t)b * S;

  // Q A-fragments (k = 0..63), reused across all key tiles.
  const _Float16* qrow = Qb + (size_t)(q0 + (lane & 15)) * D;
  v16h aq0 = load_a_frag_row(qrow, 0, lane);
  v16h aq1 = load_a_frag_row(qrow, 32, lane);

  // ---- Pass 1: online max / sum over key tiles ----
  float rm[8], rs[8];
#pragma unroll
  for (int r = 0; r < 8; ++r) {
    rm[r] = -__builtin_inff();
    rs[r] = 0.f;
  }

  for (int n0 = 0; n0 < S; n0 += 16) {
    v8f acc = vzero8();
    acc = wmma_f16(aq0, load_b_frag_rm(Ktb, 0, S, n0, lane), acc);
    acc = wmma_f16(aq1, load_b_frag_rm(Ktb, 32, S, n0, lane), acc);
    int mk = mrow[n0 + nl];
#pragma unroll
    for (int r = 0; r < 8; ++r) {
      float val = (mk == 0) ? MASK_NEG : acc[r] * scale;
      float nm = fmaxf(rm[r], val);
      rs[r] = rs[r] * __expf(rm[r] - nm) + __expf(val - nm);
      rm[r] = nm;
    }
  }
  // Merge (max,sum) across the 16 lanes that share each row.
#pragma unroll
  for (int off = 1; off < 16; off <<= 1) {
#pragma unroll
    for (int r = 0; r < 8; ++r) {
      float m2 = __shfl_xor(rm[r], off, 32);
      float s2 = __shfl_xor(rs[r], off, 32);
      float nm = fmaxf(rm[r], m2);
      rs[r] = rs[r] * __expf(rm[r] - nm) + s2 * __expf(m2 - nm);
      rm[r] = nm;
    }
  }
  float inv[8];
#pragma unroll
  for (int r = 0; r < 8; ++r) inv[r] = 1.0f / rs[r];

  // ---- Pass 2: recompute, normalize, store attn, accumulate P@V ----
  float* attn_base = attn_out + (((size_t)b * H + h) * S + q0) * S;

  v8f oacc[4];
#pragma unroll
  for (int j = 0; j < 4; ++j) oacc[j] = vzero8();

  for (int k0 = 0; k0 < S; k0 += 32) {  // pair of 16-wide key tiles
#pragma unroll
    for (int t = 0; t < 2; ++t) {
      int n0 = k0 + t * 16;
      v8f acc = vzero8();
      acc = wmma_f16(aq0, load_b_frag_rm(Ktb, 0, S, n0, lane), acc);
      acc = wmma_f16(aq1, load_b_frag_rm(Ktb, 32, S, n0, lane), acc);
      int mk = mrow[n0 + nl];
#pragma unroll
      for (int r = 0; r < 8; ++r) {
        float val = (mk == 0) ? MASK_NEG : acc[r] * scale;
        float p = __expf(val - rm[r]) * inv[r];
        attn_base[(size_t)(mb + r) * S + n0 + nl] = p;
        lp[(mb + r) * 32 + t * 16 + nl] = (_Float16)p;
      }
    }
    __syncthreads();  // P tile visible across lanes

    v16h ap;
    {
      const _Float16* lrow = lp + (lane & 15) * 32;
#pragma unroll
      for (int v = 0; v < 8; ++v) {
        int k = a_k_of(lane, v);
        ap[2 * v]     = lrow[k];
        ap[2 * v + 1] = lrow[k + 1];
      }
    }
#pragma unroll
    for (int j = 0; j < 4; ++j) {
      v16h bv = load_b_frag_rm(Vb, k0, D, j * 16, lane);
      oacc[j] = wmma_f16(ap, bv, oacc[j]);
    }
    __syncthreads();  // protect lp against next iteration's writes
  }

  _Float16* AVb = AV + (size_t)b * S * D + h * Dh;
  const int mbase = q0 + mb;
#pragma unroll
  for (int j = 0; j < 4; ++j) {
    int nc = j * 16 + nl;
#pragma unroll
    for (int r = 0; r < 8; ++r)
      AVb[(size_t)(mbase + r) * D + nc] = (_Float16)oacc[j][r];
  }
}

// ---------------------------------------------------------------------------
extern "C" void kernel_launch(void* const* d_in, const int* in_sizes, int n_in,
                              void* d_out, int out_size, void* d_ws,
                              size_t ws_size, hipStream_t stream) {
  (void)in_sizes; (void)n_in; (void)out_size; (void)ws_size;

  const int B = 2, S = 2048, D = 1024, H = 16, Dh = 64;
  const float SCALE = 0.125f;  // Dh^-0.5

  const float* x  = (const float*)d_in[0];
  const int* mask = (const int*)d_in[1];
  const float* wq = (const float*)d_in[2];
  const float* wk = (const float*)d_in[3];
  const float* wv = (const float*)d_in[4];
  const float* wo = (const float*)d_in[5];

  const size_t nBSD = (size_t)B * S * D;  // 4,194,304
  const size_t nW = (size_t)D * D;        // 1,048,576

  _Float16* wq16 = (_Float16*)d_ws;
  _Float16* wk16 = wq16 + nW;
  _Float16* wv16 = wk16 + nW;
  _Float16* wo16 = wv16 + nW;
  _Float16* Q16  = wo16 + nW;   // row-major (B*S, D), RoPE'd in place
  _Float16* Krm  = Q16 + nBSD;  // row-major K projection (pre-RoPE)
  _Float16* Kt16 = Krm + nBSD;  // transposed per head [b][h][d][s], RoPE'd
  _Float16* V16  = Kt16 + nBSD; // row-major (B*S, D)
  _Float16* AV16 = V16 + nBSD;  // row-major (B*S, D)

  float* out0 = (float*)d_out;
  float* attn = out0 + nBSD;

  // 1) weights f32 -> f16
  int tc = 256;
  cvt_f32_f16<<<((int)nW + tc - 1) / tc, tc, 0, stream>>>(wq, wq16, (int)nW);
  cvt_f32_f16<<<((int)nW + tc - 1) / tc, tc, 0, stream>>>(wk, wk16, (int)nW);
  cvt_f32_f16<<<((int)nW + tc - 1) / tc, tc, 0, stream>>>(wv, wv16, (int)nW);
  cvt_f32_f16<<<((int)nW + tc - 1) / tc, tc, 0, stream>>>(wo, wo16, (int)nW);

  // 2) projections: Q/K/V = x @ w  (A read as f32, converted in-register)
  dim3 gg(D / 64, (B * S) / 16);  // (16, 256)
  wmma_gemm<float, true><<<gg, 32, 0, stream>>>(x, wq16, Q16, B * S, D, D);
  wmma_gemm<float, true><<<gg, 32, 0, stream>>>(x, wk16, Krm, B * S, D, D);
  wmma_gemm<float, true><<<gg, 32, 0, stream>>>(x, wv16, V16, B * S, D, D);

  // 3) RoPE: Q in place, K -> transposed-per-head layout
  int nr = B * S * H * (Dh / 2);
  rope_kernel<<<(nr + tc - 1) / tc, tc, 0, stream>>>(Q16, Krm, Kt16, B, S, D, H,
                                                     Dh);

  // 4) flash-style attention (register softmax, attn store, P@V)
  dim3 ga(S / 16, H, B);
  attn_kernel<<<ga, 32, 0, stream>>>(Q16, Kt16, V16, mask, attn, AV16, B, H, S,
                                     D, Dh, SCALE);

  // 5) out = AV @ wo  (f32 output)
  wmma_gemm<_Float16, false><<<gg, 32, 0, stream>>>(AV16, wo16, out0, B * S, D, D);
}
